// IterativeRefinementDecoder_39161511805375
// MI455X (gfx1250) — matrix operand order repaired and checked by reading, not compiled, not run on hardware
//
#include <hip/hip_runtime.h>
#include <hip/hip_bf16.h>

// ---------------- problem constants (match reference) ----------------
#define L_LAYERS 6
#define Bq   8
#define Nq   300
#define Cq   256
#define NHq  8
#define NPq  4
#define DHq  32
#define DFFq 1024
#define HGTq 100
#define WIDq 100
#define HWq  (HGTq * WIDq)
#define EPSq 1e-5f

typedef __attribute__((ext_vector_type(16))) __bf16 bf16x16;
typedef __attribute__((ext_vector_type(8)))  __bf16 bf16x8;
typedef __attribute__((ext_vector_type(8)))  float  floatx8;

// =====================================================================
// Generic GEMM:  C[M,N] = A[M,K] @ W[N,K]^T + bias[N]   (optional ReLU)
// f32 in / f32 out, bf16 WMMA compute with f32 accumulation.
//
// REQUIRES K % 32 == 0 (true for every GEMM in this model: K=256 or 1024).
// Block tile 128(M) x 128(N), K-step 32, 8 waves; wave tile 32x64 =
// 2x4 of v_wmma_f32_16x16x32_bf16 per K-step.
// Branchless staging: row/col indices are clamped (OOB rows/cols feed
// accumulator lanes that are never stored), so global loads are b128
// and LDS traffic is b128 (A) / packed b32 (W-transpose).
// =====================================================================
__global__ __launch_bounds__(256)
void gemm_bf16_wmma(const float* __restrict__ A,
                    const float* __restrict__ Wm,    // [N,K] row-major
                    const float* __restrict__ bias,  // [N]
                    float* __restrict__ C,
                    int M, int N, int K, int relu)
{
  __shared__ __bf16 sA[128][40];    // [m][k]   row stride 80B (16B-aligned)
  __shared__ __bf16 sBT[32][136];   // [k][n]   row stride 272B (16B-aligned)

  const int tid  = threadIdx.x;
  const int lane = tid & 31;
  const int wave = tid >> 5;
  const int wy   = wave >> 1;   // 0..3  -> M sub-tile of 32
  const int wx   = wave & 1;    // 0..1  -> N sub-tile of 64
  const int m0   = blockIdx.y * 128;
  const int n0   = blockIdx.x * 128;

  // ---- staging coordinates (clamped -> branchless) ----
  // A: thread covers one row, 16 k's
  const int ar  = tid >> 1;                    // 0..127 : A tile row
  const int ac0 = (tid & 1) * 16;              // 0 / 16 : A k-offset
  const int gm  = min(m0 + ar, M - 1);
  // W: thread covers a pair of n-columns for 8 k's (packed b32 stores)
  const int bp  = tid >> 2;                    // 0..63  : n-pair index
  const int bkg = (tid & 3) * 8;               // 0/8/16/24 : k-offset
  const int gn0 = min(n0 + 2 * bp,     N - 1);
  const int gn1 = min(n0 + 2 * bp + 1, N - 1);

  const float* __restrict__ aRow  = A  + (size_t)gm  * K;
  const float* __restrict__ wRow0 = Wm + (size_t)gn0 * K;
  const float* __restrict__ wRow1 = Wm + (size_t)gn1 * K;

  floatx8 acc[2][4];
#pragma unroll
  for (int i = 0; i < 2; ++i)
#pragma unroll
    for (int j = 0; j < 4; ++j)
      acc[i][j] = (floatx8){0.f, 0.f, 0.f, 0.f, 0.f, 0.f, 0.f, 0.f};

  for (int k0 = 0; k0 < K; k0 += 32) {
    // ---- stage A tile 128x32 : 4x b128 loads, 2x b128 LDS stores ----
    {
      const float4* src = (const float4*)(aRow + k0 + ac0);
      float4 f0 = src[0], f1 = src[1], f2 = src[2], f3 = src[3];
      if (k0 + 32 < K)  // pull next k-tile toward the WGP (global_prefetch_b8)
        __builtin_prefetch((const void*)(aRow + k0 + 32 + ac0), 0, 3);
      bf16x8 lo, hi;
      lo[0] = (__bf16)f0.x; lo[1] = (__bf16)f0.y; lo[2] = (__bf16)f0.z; lo[3] = (__bf16)f0.w;
      lo[4] = (__bf16)f1.x; lo[5] = (__bf16)f1.y; lo[6] = (__bf16)f1.z; lo[7] = (__bf16)f1.w;
      hi[0] = (__bf16)f2.x; hi[1] = (__bf16)f2.y; hi[2] = (__bf16)f2.z; hi[3] = (__bf16)f2.w;
      hi[4] = (__bf16)f3.x; hi[5] = (__bf16)f3.y; hi[6] = (__bf16)f3.z; hi[7] = (__bf16)f3.w;
      *(bf16x8*)&sA[ar][ac0]     = lo;
      *(bf16x8*)&sA[ar][ac0 + 8] = hi;
    }
    // ---- stage W tile 128x32 TRANSPOSED: 4x b128 loads, 8x packed b32 ----
    {
      const float4* s0 = (const float4*)(wRow0 + k0 + bkg);
      const float4* s1 = (const float4*)(wRow1 + k0 + bkg);
      float4 a0 = s0[0], a1 = s0[1];
      float4 b0 = s1[0], b1 = s1[1];
      float r0[8] = {a0.x, a0.y, a0.z, a0.w, a1.x, a1.y, a1.z, a1.w};
      float r1[8] = {b0.x, b0.y, b0.z, b0.w, b1.x, b1.y, b1.z, b1.w};
#pragma unroll
      for (int i = 0; i < 8; ++i) {
        union { __bf16 h[2]; unsigned u; } pk;
        pk.h[0] = (__bf16)r0[i];   // column n = 2*bp
        pk.h[1] = (__bf16)r1[i];   // column n = 2*bp+1
        *(unsigned*)&sBT[bkg + i][2 * bp] = pk.u;   // 4B-aligned ds_store_b32
      }
    }
    __syncthreads();

    // ---- fragments (CDNA5 16-bit WMMA layouts) ----
    // A 16x32: lanes 0-15 -> M=lane, K 0..7 & 16..23 ; lanes 16-31 -> K 8..15 & 24..31
    const int m_ = lane & 15;
    const int kh = lane >> 4;
    bf16x16 afrag[2], bfrag[4];
#pragma unroll
    for (int mt = 0; mt < 2; ++mt) {
      const int rb = wy * 32 + mt * 16 + m_;
      union { bf16x16 v; bf16x8 h[2]; } u;
      u.h[0] = *(const bf16x8*)&sA[rb][kh * 8];
      u.h[1] = *(const bf16x8*)&sA[rb][16 + kh * 8];
      afrag[mt] = u.v;
    }
    // B 32x16: lanes 0-15 -> N=lane, K 0..7 & 16..23 ; lanes 16-31 -> K 8..15 & 24..31
#pragma unroll
    for (int nt = 0; nt < 4; ++nt) {
      const int nb = wx * 64 + nt * 16 + m_;
      bf16x16 v;
#pragma unroll
      for (int s = 0; s < 8; ++s) {
        v[s]     = sBT[kh * 8 + s][nb];
        v[8 + s] = sBT[16 + kh * 8 + s][nb];
      }
      bfrag[nt] = v;
    }
#pragma unroll
    for (int mt = 0; mt < 2; ++mt)
#pragma unroll
      for (int nt = 0; nt < 4; ++nt)
        acc[mt][nt] = __builtin_amdgcn_wmma_f32_16x16x32_bf16(
            false, afrag[mt], false, bfrag[nt], (short)0, acc[mt][nt],
            false, false);
    __syncthreads();
  }

  // ---- writeout: C/D layout lane n = lane&15, M = (lane>>4)*8 + v ----
#pragma unroll
  for (int mt = 0; mt < 2; ++mt) {
#pragma unroll
    for (int nt = 0; nt < 4; ++nt) {
      const int col = n0 + wx * 64 + nt * 16 + (lane & 15);
      if (col >= N) continue;
      const float bcol = bias ? bias[col] : 0.0f;
#pragma unroll
      for (int v = 0; v < 8; ++v) {
        const int row = m0 + wy * 32 + mt * 16 + (lane >> 4) * 8 + v;
        if (row < M) {
          float o = acc[mt][nt][v] + bcol;
          if (relu) o = fmaxf(o, 0.0f);
          C[(size_t)row * N + col] = o;
        }
      }
    }
  }
}

// =====================================================================
// MHA: one wave per (b, h, query-row). qkv layout [B,N,768].
// =====================================================================
__global__ __launch_bounds__(256)
void attn_kernel(const float* __restrict__ qkv, float* __restrict__ out)
{
  const int bh   = blockIdx.x;          // 0..B*NH-1
  const int b    = bh >> 3;
  const int h    = bh & 7;
  const int wave = threadIdx.x >> 5;
  const int lane = threadIdx.x & 31;
  const int n    = blockIdx.y * 8 + wave;
  const bool valid = (n < Nq);
  const int  nc    = valid ? n : (Nq - 1);

  __shared__ float sq[8][32];
  __shared__ float satt[8][304];

  sq[wave][lane] = qkv[((size_t)b * Nq + nc) * 768 + h * 32 + lane];
  __syncthreads();

  const float scale = 0.17677669529663687f;  // 1/sqrt(32)
  float lmax = -3.0e38f;
  for (int j = lane; j < Nq; j += 32) {
    const float* kp = qkv + ((size_t)b * Nq + j) * 768 + 256 + h * 32;
    float s = 0.f;
#pragma unroll
    for (int d = 0; d < 32; ++d) s += sq[wave][d] * kp[d];
    s *= scale;
    satt[wave][j] = s;
    lmax = fmaxf(lmax, s);
  }
#pragma unroll
  for (int off = 16; off > 0; off >>= 1)
    lmax = fmaxf(lmax, __shfl_xor(lmax, off, 32));

  float lsum = 0.f;
  for (int j = lane; j < Nq; j += 32) {
    const float e = __expf(satt[wave][j] - lmax);
    satt[wave][j] = e;
    lsum += e;
  }
#pragma unroll
  for (int off = 16; off > 0; off >>= 1)
    lsum += __shfl_xor(lsum, off, 32);
  __syncthreads();

  const float inv = 1.0f / lsum;
  float o = 0.f;
  for (int j = 0; j < Nq; ++j)
    o += satt[wave][j] * qkv[((size_t)b * Nq + j) * 768 + 512 + h * 32 + lane];
  if (valid)
    out[((size_t)b * Nq + n) * Cq + h * 32 + lane] = o * inv;
}

// =====================================================================
// y = LN(x + r) * g + b     (block of 256 per row; C == 256)
// =====================================================================
__global__ __launch_bounds__(256)
void ln_kernel(const float* __restrict__ x, const float* __restrict__ r,
               const float* __restrict__ g, const float* __restrict__ be,
               float* __restrict__ y, float* __restrict__ y2)
{
  const int row = blockIdx.x;
  const int t   = threadIdx.x;
  __shared__ float red[256];
  const size_t idx = (size_t)row * Cq + t;
  const float v = x[idx] + r[idx];
  red[t] = v;
  __syncthreads();
  for (int s = 128; s > 0; s >>= 1) {
    if (t < s) red[t] += red[t + s];
    __syncthreads();
  }
  const float mean = red[0] * (1.0f / Cq);
  __syncthreads();
  const float d = v - mean;
  red[t] = d * d;
  __syncthreads();
  for (int s = 128; s > 0; s >>= 1) {
    if (t < s) red[t] += red[t + s];
    __syncthreads();
  }
  const float var = red[0] * (1.0f / Cq);
  const float o = d * rsqrtf(var + EPSq) * g[t] + be[t];
  y[idx] = o;
  if (y2) y2[idx] = o;
}

// =====================================================================
// loc[idx] = (ref[b,n,c] + tanh(off_lin[idx])*0.5)*2 - 1
// (flat index matches the reference's torch-faithful reshape)
// =====================================================================
__global__ void loc_kernel(const float* __restrict__ off_lin,
                           const float* __restrict__ refb,
                           float* __restrict__ loc)
{
  const int idx = blockIdx.x * blockDim.x + threadIdx.x;
  if (idx >= Bq * Nq * NHq * NPq * 2) return;
  const int bn = idx >> 6;  // / (NH*NP*2)
  const int c  = idx & 1;
  const float off = tanhf(off_lin[idx]) * 0.5f;
  loc[idx] = (refb[bn * 2 + c] + off) * 2.0f - 1.0f;
}

// softmax over NP=4 per (b,n,h)
__global__ void aw_kernel(const float* __restrict__ awl, float* __restrict__ aw)
{
  const int idx = blockIdx.x * blockDim.x + threadIdx.x;
  if (idx >= Bq * Nq * NHq) return;
  const float* s = awl + (size_t)idx * 4;
  const float m = fmaxf(fmaxf(s[0], s[1]), fmaxf(s[2], s[3]));
  float e0 = __expf(s[0] - m), e1 = __expf(s[1] - m);
  float e2 = __expf(s[2] - m), e3 = __expf(s[3] - m);
  const float inv = 1.0f / (e0 + e1 + e2 + e3);
  float* d = aw + (size_t)idx * 4;
  d[0] = e0 * inv; d[1] = e1 * inv; d[2] = e2 * inv; d[3] = e3 * inv;
}

// =====================================================================
// Deformable sampling: block per (b,n); thread = (h,d).
// Reproduces the [B,N,NH,NP,2] -> [B*NH, N*NP, 2] flat-reshape quirk:
// location for output (b,n,h,p) is loc[b][ h*N*NP + n*NP + p ].
// =====================================================================
__global__ __launch_bounds__(256)
void deform_kernel(const float* __restrict__ loc, const float* __restrict__ aw,
                   const float* __restrict__ val, float* __restrict__ samp)
{
  const int bn = blockIdx.x;
  const int b  = bn / Nq, n = bn % Nq;
  const int h  = threadIdx.x >> 5, d = threadIdx.x & 31;
  const float* locb = loc + (size_t)b * (Nq * NHq * NPq * 2);
  const float* vb   = val + (size_t)b * HWq * Cq + h * 32 + d;

  auto tap = [&](int xi, int yi) -> float {
    if (xi < 0 || xi >= WIDq || yi < 0 || yi >= HGTq) return 0.f;
    return vb[(size_t)(yi * WIDq + xi) * Cq];
  };

  float acc = 0.f;
#pragma unroll
  for (int p = 0; p < NPq; ++p) {
    const int lidx = (h * (Nq * NPq) + n * NPq + p) * 2;
    const float cx = locb[lidx], cy = locb[lidx + 1];
    const float x = ((cx + 1.0f) * WIDq - 1.0f) * 0.5f;
    const float y = ((cy + 1.0f) * HGTq - 1.0f) * 0.5f;
    const float x0f = floorf(x), y0f = floorf(y);
    const int x0 = (int)x0f, y0 = (int)y0f;
    const float wx1 = x - x0f, wx0 = 1.0f - wx1;
    const float wy1 = y - y0f, wy0 = 1.0f - wy1;
    const float s = tap(x0, y0) * wx0 * wy0 + tap(x0 + 1, y0) * wx1 * wy0 +
                    tap(x0, y0 + 1) * wx0 * wy1 + tap(x0 + 1, y0 + 1) * wx1 * wy1;
    acc += aw[(size_t)bn * 32 + h * 4 + p] * s;
  }
  samp[(size_t)bn * Cq + h * 32 + d] = acc;
}

// =====================================================================
// ref update: delta = sigmoid(q @ Wref^T + bref); ref = clip(ref + (d-.5)*.1)
// =====================================================================
__global__ __launch_bounds__(64)
void ref_kernel(const float* __restrict__ q, const float* __restrict__ Wref,
                const float* __restrict__ bref, float* __restrict__ refb,
                float* __restrict__ refs_out)
{
  const int bn = blockIdx.x;
  const int c = threadIdx.x >> 5, lane = threadIdx.x & 31;
  const float* qr = q + (size_t)bn * Cq;
  const float* wr = Wref + c * Cq;
  float s = 0.f;
  for (int i = lane; i < Cq; i += 32) s += qr[i] * wr[i];
#pragma unroll
  for (int off = 16; off > 0; off >>= 1) s += __shfl_xor(s, off, 32);
  if (lane == 0) {
    const float delta = 1.0f / (1.0f + __expf(-(s + bref[c])));
    float r = refb[bn * 2 + c] + (delta - 0.5f) * 0.1f;
    r = fminf(fmaxf(r, 0.0f), 1.0f);
    refb[bn * 2 + c] = r;
    refs_out[bn * 2 + c] = r;
  }
}

__global__ void init_kernel(const float* __restrict__ query, float* __restrict__ q,
                            float* __restrict__ refb, float* __restrict__ refs0)
{
  const int idx = blockIdx.x * blockDim.x + threadIdx.x;
  if (idx < Bq * Nq * Cq) q[idx] = query[idx];
  if (idx < Bq * Nq * 2) { refb[idx] = 0.5f; refs0[idx] = 0.5f; }
}

// =====================================================================
extern "C" void kernel_launch(void* const* d_in, const int* in_sizes, int n_in,
                              void* d_out, int out_size, void* d_ws, size_t ws_size,
                              hipStream_t stream)
{
  const float* query  = (const float*)d_in[0];
  const float* memory = (const float*)d_in[1];
  const float* Wqkv = (const float*)d_in[2];
  const float* bqkv = (const float*)d_in[3];
  const float* Wo   = (const float*)d_in[4];
  const float* bo   = (const float*)d_in[5];
  const float* Woff = (const float*)d_in[6];
  const float* boff = (const float*)d_in[7];
  const float* Waw  = (const float*)d_in[8];
  const float* baw  = (const float*)d_in[9];
  const float* Wv   = (const float*)d_in[10];
  const float* bv   = (const float*)d_in[11];
  const float* Wc   = (const float*)d_in[12];
  const float* bc   = (const float*)d_in[13];
  const float* W1   = (const float*)d_in[14];
  const float* b1   = (const float*)d_in[15];
  const float* W2m  = (const float*)d_in[16];
  const float* b2   = (const float*)d_in[17];
  const float* g1   = (const float*)d_in[18];
  const float* be1  = (const float*)d_in[19];
  const float* g2   = (const float*)d_in[20];
  const float* be2  = (const float*)d_in[21];
  const float* g3   = (const float*)d_in[22];
  const float* be3  = (const float*)d_in[23];
  const float* Wref = (const float*)d_in[24];
  const float* bref = (const float*)d_in[25];
  (void)in_sizes; (void)n_in; (void)out_size; (void)ws_size;

  float* outs = (float*)d_out;                                  // [L,B,N,C]
  float* refs = outs + (size_t)L_LAYERS * Bq * Nq * Cq;         // [L+1,B,N,2]

  // ---- workspace carve-up (f32), ~105 MB total ----
  float* ws = (float*)d_ws;
  size_t o = 0;
  auto carve = [&](size_t nelem) { float* p = ws + o; o += nelem; return p; };
  float* qbuf = carve((size_t)Bq * Nq * Cq);
  float* refb = carve((size_t)Bq * Nq * 2);
  float* qkv  = carve((size_t)Bq * Nq * 3 * Cq);
  float* att  = carve((size_t)Bq * Nq * Cq);
  float* tmp  = carve((size_t)Bq * Nq * Cq);
  float* val  = carve((size_t)Bq * HWq * Cq);   // 82 MB — L2-resident on MI455X
  float* offl = carve((size_t)Bq * Nq * 64);
  float* awl  = carve((size_t)Bq * Nq * 32);
  float* awb  = carve((size_t)Bq * Nq * 32);
  float* locb = carve((size_t)Bq * Nq * 64);
  float* h1b  = carve((size_t)Bq * Nq * DFFq);
  float* samp = carve((size_t)Bq * Nq * Cq);

  const int M = Bq * Nq;  // 2400

  auto gemm = [&](const float* A, const float* Wmat, const float* bias, float* C,
                  int Mm, int Nn, int Kk, int relu) {
    dim3 g((Nn + 127) / 128, (Mm + 127) / 128);
    gemm_bf16_wmma<<<g, dim3(256), 0, stream>>>(A, Wmat, bias, C, Mm, Nn, Kk, relu);
  };

  init_kernel<<<(M * Cq + 255) / 256, 256, 0, stream>>>(query, qbuf, refb, refs);

  for (int l = 0; l < L_LAYERS; ++l) {
    // --- self attention ---
    gemm(qbuf, Wqkv + (size_t)l * 3 * Cq * Cq, bqkv + (size_t)l * 3 * Cq,
         qkv, M, 3 * Cq, Cq, 0);
    attn_kernel<<<dim3(Bq * NHq, (Nq + 7) / 8), 256, 0, stream>>>(qkv, att);
    gemm(att, Wo + (size_t)l * Cq * Cq, bo + (size_t)l * Cq, tmp, M, Cq, Cq, 0);
    ln_kernel<<<M, 256, 0, stream>>>(tmp, qbuf, g1 + (size_t)l * Cq,
                                     be1 + (size_t)l * Cq, qbuf, (float*)nullptr);
    // --- deformable cross attention ---
    gemm(memory, Wv + (size_t)l * Cq * Cq, bv + (size_t)l * Cq,
         val, Bq * HWq, Cq, Cq, 0);
    gemm(qbuf, Woff + (size_t)l * 64 * Cq, boff + (size_t)l * 64, offl, M, 64, Cq, 0);
    gemm(qbuf, Waw + (size_t)l * 32 * Cq, baw + (size_t)l * 32, awl, M, 32, Cq, 0);
    loc_kernel<<<(M * 64 + 255) / 256, 256, 0, stream>>>(offl, refb, locb);
    aw_kernel<<<(M * 8 + 255) / 256, 256, 0, stream>>>(awl, awb);
    deform_kernel<<<M, 256, 0, stream>>>(locb, awb, val, samp);
    gemm(samp, Wc + (size_t)l * Cq * Cq, bc + (size_t)l * Cq, tmp, M, Cq, Cq, 0);
    ln_kernel<<<M, 256, 0, stream>>>(tmp, qbuf, g2 + (size_t)l * Cq,
                                     be2 + (size_t)l * Cq, qbuf, (float*)nullptr);
    // --- FFN ---
    gemm(qbuf, W1 + (size_t)l * DFFq * Cq, b1 + (size_t)l * DFFq, h1b, M, DFFq, Cq, 1);
    gemm(h1b, W2m + (size_t)l * Cq * DFFq, b2 + (size_t)l * Cq, tmp, M, Cq, DFFq, 0);
    ln_kernel<<<M, 256, 0, stream>>>(tmp, qbuf, g3 + (size_t)l * Cq,
                                     be3 + (size_t)l * Cq, qbuf,
                                     outs + (size_t)l * M * Cq);
    // --- ref point refinement ---
    ref_kernel<<<M, 64, 0, stream>>>(qbuf, Wref + (size_t)l * 2 * Cq,
                                     bref + (size_t)l * 2, refb,
                                     refs + (size_t)(l + 1) * M * 2);
  }
}
